// LSTM_17205638987779
// MI455X (gfx1250) — compile-verified, matrix operand order checked by reading, not run on hardware
//
#include <hip/hip_runtime.h>
#include <hip/hip_bf16.h>

// ---------------------------------------------------------------------------
// LSTM cell step on gfx1250 via bf16 WMMA + async global->LDS tile DMA.
//   A  = [x | h | c]                  : [16384 x 2560] bf16 (packed)
//   Bt = gate-interleaved weights^T   : [4096  x 2560] bf16 (packed)
//       n = 64*hb + 16*gate + hl ; hcol = 16*hb + hl
//       k in [0,512)    -> W_gate[k][hcol]
//       k in [512,1536) -> U_gate[k-512][hcol]
//       k in [1536,2560)-> gate==3 ? o_V[k-1536][hcol] : 0   (peephole term)
//   z = A @ Bt^T accumulated fp32 in WMMA fragments, gates fused in epilogue.
//   Tile fill uses GLOBAL_LOAD_ASYNC_TO_LDS_B128 (ASYNCcnt) - no VGPR staging.
// ---------------------------------------------------------------------------

#define B_SZ   16384
#define I_SZ   512
#define H_SZ   1024
#define KD     2560      // 512 + 1024 + 1024
#define ND     4096      // 1024 h cols * 4 gates (interleaved)
#define KT     32        // K per bf16 WMMA
#define LDA    40        // padded LDS row stride (bf16 elems) = 80B
#define MT     256       // workgroup M tile
#define NT     128       // workgroup N tile

typedef __bf16 bf16;
typedef __attribute__((ext_vector_type(16))) __bf16 v16bf;
typedef __attribute__((ext_vector_type(8)))  float  v8f;

union Frag { v16bf v; uint4 u[2]; };

__device__ __forceinline__ void async_b128(unsigned lds_off, unsigned long long gaddr) {
    // GLOBAL_LOAD_ASYNC_TO_LDS_B128: copies 16B from global vaddr into LDS at vdst, tracked by ASYNCcnt
    asm volatile("global_load_async_to_lds_b128 %0, %1, off"
                 :: "v"(lds_off), "v"(gaddr) : "memory");
}
__device__ __forceinline__ void wait_async0() {
    asm volatile("s_wait_asynccnt 0x0" ::: "memory");
}

// ---------------------------------------------------------------- pack A ----
__global__ void pack_A(const float* __restrict__ x, const float* __restrict__ h,
                       const float* __restrict__ c, bf16* __restrict__ A) {
    size_t idx = (size_t)blockIdx.x * blockDim.x + threadIdx.x;
    if (idx >= (size_t)B_SZ * KD) return;
    int    k   = (int)(idx % KD);
    size_t row = idx / KD;
    float v;
    if (k < I_SZ)              v = x[row * I_SZ + k];
    else if (k < I_SZ + H_SZ)  v = h[row * H_SZ + (k - I_SZ)];
    else                       v = c[row * H_SZ + (k - I_SZ - H_SZ)];
    A[idx] = (bf16)v;
}

// ---------------------------------------------------------------- pack B ----
__global__ void pack_B(const float* __restrict__ iW, const float* __restrict__ fW,
                       const float* __restrict__ gW, const float* __restrict__ oW,
                       const float* __restrict__ iU, const float* __restrict__ fU,
                       const float* __restrict__ gU, const float* __restrict__ oU,
                       const float* __restrict__ oV, bf16* __restrict__ Bt) {
    size_t idx = (size_t)blockIdx.x * blockDim.x + threadIdx.x;
    if (idx >= (size_t)ND * KD) return;
    int k = (int)(idx % KD);
    int n = (int)(idx / KD);
    int hb = n >> 6, rem = n & 63, gate = rem >> 4, hl = rem & 15;
    int hcol = hb * 16 + hl;
    float v = 0.0f;
    if (k < I_SZ) {
        const float* W = (gate == 0) ? iW : (gate == 1) ? fW : (gate == 2) ? gW : oW;
        v = W[(size_t)k * H_SZ + hcol];
    } else if (k < I_SZ + H_SZ) {
        const float* U = (gate == 0) ? iU : (gate == 1) ? fU : (gate == 2) ? gU : oU;
        v = U[(size_t)(k - I_SZ) * H_SZ + hcol];
    } else if (gate == 3) {
        v = oV[(size_t)(k - I_SZ - H_SZ) * H_SZ + hcol];
    }
    Bt[idx] = (bf16)v;
}

// --------------------------------------------------------------- init out ---
__global__ void init_out(float* __restrict__ out, const float* __restrict__ out_b) {
    int i = blockIdx.x * blockDim.x + threadIdx.x;
    if (i < B_SZ) out[i] = out_b[0];
}

// ------------------------------------------------------ fused WMMA GEMM -----
__global__ __launch_bounds__(256)
void lstm_gemm(const bf16* __restrict__ A, const bf16* __restrict__ Bt,
               const float* __restrict__ c_prev,
               const float* __restrict__ i_b, const float* __restrict__ f_b,
               const float* __restrict__ g_b, const float* __restrict__ o_b,
               const float* __restrict__ outW,
               float* __restrict__ out)   // [0,B): output ; then h_t ; then c_t
{
    __shared__ __align__(16) bf16 As[2][MT * LDA];   // 2 x 20 KiB
    __shared__ __align__(16) bf16 Bs[2][NT * LDA];   // 2 x 10 KiB

    const int tid  = threadIdx.x;
    const int lane = tid & 31;
    const int wid  = tid >> 5;
    const int wm   = wid & 3;      // 4 M-groups of 64 rows
    const int wn   = wid >> 2;     // 2 N-groups of 64 cols
    const int hl   = lane & 15;
    const int half = lane >> 4;
    const int mbase = blockIdx.y * MT;
    const int nbase = blockIdx.x * NT;

    v8f acc[4][4];
    const v8f vzero = {0.f, 0.f, 0.f, 0.f, 0.f, 0.f, 0.f, 0.f};
#pragma unroll
    for (int mi = 0; mi < 4; ++mi)
#pragma unroll
        for (int g = 0; g < 4; ++g) acc[mi][g] = vzero;

    // Per-thread async-copy slots: A tile = 256 rows x 4 16B chunks = 1024,
    // B tile = 128 rows x 4 chunks = 512 -> 4 + 2 chunks per thread per step.
    unsigned long long gA[4], gB[2];
    unsigned lA[4], lB[2];
#pragma unroll
    for (int i = 0; i < 4; ++i) {
        int j = tid + i * 256, row = j >> 2, kc = j & 3;
        gA[i] = (unsigned long long)(A + (size_t)(mbase + row) * KD + kc * 8);
        lA[i] = (unsigned)(uintptr_t)(&As[0][row * LDA + kc * 8]);
    }
#pragma unroll
    for (int i = 0; i < 2; ++i) {
        int j = tid + i * 256, row = j >> 2, kc = j & 3;
        gB[i] = (unsigned long long)(Bt + (size_t)(nbase + row) * KD + kc * 8);
        lB[i] = (unsigned)(uintptr_t)(&Bs[0][row * LDA + kc * 8]);
    }
    const unsigned aStride = MT * LDA * 2;   // bytes between A buffers
    const unsigned bStride = NT * LDA * 2;   // bytes between B buffers

    auto issue_tile = [&](int buf) {
#pragma unroll
        for (int i = 0; i < 4; ++i) { async_b128(lA[i] + buf * aStride, gA[i]); gA[i] += KT * 2; }
#pragma unroll
        for (int i = 0; i < 2; ++i) { async_b128(lB[i] + buf * bStride, gB[i]); gB[i] += KT * 2; }
    };

    issue_tile(0);
    wait_async0();
    __syncthreads();

    const int KSTEPS = KD / KT;               // 80
    for (int kt = 0; kt < KSTEPS; ++kt) {
        const int cur = kt & 1;
        if (kt + 1 < KSTEPS) issue_tile(cur ^ 1);

        // 16-bit fragment layout (ISA 7.12.2): lane m/n = hl,
        // half 0 -> K 0..7 & 16..23, half 1 -> K 8..15 & 24..31, i.e. two
        // contiguous 16B chunks at byte offsets half*16 and half*16+32.
        Frag b[4];
#pragma unroll
        for (int g = 0; g < 4; ++g) {
            const bf16* p = &Bs[cur][(wn * 64 + g * 16 + hl) * LDA + half * 8];
            b[g].u[0] = *(const uint4*)(p);
            b[g].u[1] = *(const uint4*)(p + 16);
        }
#pragma unroll
        for (int mi = 0; mi < 4; ++mi) {
            Frag a;
            const bf16* p = &As[cur][(wm * 64 + mi * 16 + hl) * LDA + half * 8];
            a.u[0] = *(const uint4*)(p);
            a.u[1] = *(const uint4*)(p + 16);
#pragma unroll
            for (int g = 0; g < 4; ++g)
                acc[mi][g] = __builtin_amdgcn_wmma_f32_16x16x32_bf16(
                    false, a.v, false, b[g].v, (short)0, acc[mi][g], false, false);
        }

        wait_async0();     // next tile fully in LDS before anyone reuses buffers
        __syncthreads();
    }

    // ---- fused LSTM epilogue (all four gates are in-lane) -----------------
    const int hb   = blockIdx.x * 2 + wn;     // 16-wide h block
    const int hcol = hb * 16 + hl;
    const float bi = i_b[hcol], bfv = f_b[hcol], bg = g_b[hcol], bo = o_b[hcol];
    const float w_out = outW[hcol];           // out_W is [H,1]
    float* __restrict__ h_out = out + B_SZ;
    float* __restrict__ c_out = out + B_SZ + (size_t)B_SZ * H_SZ;

#pragma unroll
    for (int mi = 0; mi < 4; ++mi) {
#pragma unroll
        for (int r = 0; r < 8; ++r) {
            const int row = mbase + wm * 64 + mi * 16 + half * 8 + r;
            const float zi = acc[mi][0][r] + bi;
            const float zf = acc[mi][1][r] + bfv;
            const float zg = acc[mi][2][r] + bg;
            const float zo = acc[mi][3][r] + bo;   // already includes c @ o_V
            const float it = 1.0f / (1.0f + __expf(-zi));
            const float ft = 1.0f / (1.0f + __expf(-zf));
            const float gt = 1.0f / (1.0f + __expf(-zg));   // reference uses sigmoid
            const float ot = 1.0f / (1.0f + __expf(-zo));
            const float cp = c_prev[(size_t)row * H_SZ + hcol];
            const float ct = it * gt + ft * cp;
            const float e2 = __expf(2.0f * ct);             // tanh via exp
            const float th = 1.0f - 2.0f / (e2 + 1.0f);
            const float ht = ot * th;
            h_out[(size_t)row * H_SZ + hcol] = ht;
            c_out[(size_t)row * H_SZ + hcol] = ct;

            // partial of h_t @ out_W over this wave's 16 h columns
            float partial = ht * w_out;
#pragma unroll
            for (int off = 8; off > 0; off >>= 1)
                partial += __shfl_xor(partial, off, 32);   // stays inside half-group
            if (hl == 0) atomicAdd(&out[row], partial);
        }
    }
}

// ---------------------------------------------------------------------------
extern "C" void kernel_launch(void* const* d_in, const int* in_sizes, int n_in,
                              void* d_out, int out_size, void* d_ws, size_t ws_size,
                              hipStream_t stream) {
    const float* x    = (const float*)d_in[0];
    const float* h    = (const float*)d_in[1];
    const float* c    = (const float*)d_in[2];
    const float* iW   = (const float*)d_in[3];
    const float* iU   = (const float*)d_in[4];
    const float* ib   = (const float*)d_in[5];
    const float* fW   = (const float*)d_in[6];
    const float* fU   = (const float*)d_in[7];
    const float* fb   = (const float*)d_in[8];
    const float* gW   = (const float*)d_in[9];
    const float* gU   = (const float*)d_in[10];
    const float* gb   = (const float*)d_in[11];
    const float* oW   = (const float*)d_in[12];
    const float* oU   = (const float*)d_in[13];
    const float* oV   = (const float*)d_in[14];
    const float* ob   = (const float*)d_in[15];
    const float* outW = (const float*)d_in[16];
    const float* outb = (const float*)d_in[17];
    float* out = (float*)d_out;

    bf16* Abf = (bf16*)d_ws;                       // 16384*2560*2 = 80 MiB
    bf16* Btb = Abf + (size_t)B_SZ * KD;           // 4096*2560*2  = 20 MiB

    {   size_t tot = (size_t)B_SZ * KD;
        pack_A<<<(unsigned)((tot + 255) / 256), 256, 0, stream>>>(x, h, c, Abf); }
    {   size_t tot = (size_t)ND * KD;
        pack_B<<<(unsigned)((tot + 255) / 256), 256, 0, stream>>>(
            iW, fW, gW, oW, iU, fU, gU, oU, oV, Btb); }
    init_out<<<(B_SZ + 255) / 256, 256, 0, stream>>>(out, outb);

    dim3 grid(ND / NT, B_SZ / MT);                 // 32 x 64 workgroups
    lstm_gemm<<<grid, 256, 0, stream>>>(Abf, Btb, c, ib, fb, gb, ob, outW, out);
}
